// CrossAttentionBirchSan_63823214019167
// MI455X (gfx1250) — matrix operand order repaired and checked by reading, not compiled, not run on hardware
//
#include <hip/hip_runtime.h>

// ---------------------------------------------------------------------------
// CrossAttention forward on MI455X (gfx1250).
// All matmuls via v_wmma_f32_16x16x32_bf16 (wave32), f32 accum, f32 softmax.
// B-tiles staged into LDS with the Tensor Data Mover (tensor_load_to_lds,
// double-buffered, s_wait_tensorcnt) when the builtin is available.
// ---------------------------------------------------------------------------

#define BB 2
#define SS 2048
#define DD 1024
#define HH 16
#define DHD 64             // dim head
#define MTOT (BB*SS)       // 4096
#define NPROJ (HH*DHD)     // 1024
#define ATTN_SCALE 0.125f  // 64^-0.5

typedef __bf16 bf16;
typedef __attribute__((ext_vector_type(16))) __bf16 bf16x16;
typedef __attribute__((ext_vector_type(8)))  __bf16 bf16x8;
typedef __attribute__((ext_vector_type(8)))  float  f32x8;
typedef __attribute__((ext_vector_type(4)))  unsigned int u32x4;
typedef __attribute__((ext_vector_type(8)))  int    i32x8;
typedef __attribute__((ext_vector_type(4)))  int    i32x4;

#ifndef __has_builtin
#define __has_builtin(x) 0
#endif
#if __has_builtin(__builtin_amdgcn_tensor_load_to_lds)
#define HAVE_TDM 1
#else
#define HAVE_TDM 0
#endif

#if __has_builtin(__builtin_amdgcn_s_wait_tensorcnt)
#define WAIT_TENSORCNT(n) __builtin_amdgcn_s_wait_tensorcnt(n)
#else
#define WAIT_TENSORCNT(n) asm volatile("s_wait_tensorcnt %0" :: "i"(n) : "memory")
#endif

union Frag16 { bf16x16 v; bf16x8 h[2]; };

__device__ __forceinline__ f32x8 wmma_bf16(bf16x16 a, bf16x16 b, f32x8 c) {
    // (neg_a, A, neg_b, B, c_mod, C, reuse_a, reuse_b)
    return __builtin_amdgcn_wmma_f32_16x16x32_bf16(
        false, a, false, b, (short)0, c, false, false);
}

// ---------------------------------------------------------------------------
// fp32 -> bf16 convert (straight)
// ---------------------------------------------------------------------------
__global__ void cvt_f32_bf16(const float* __restrict__ in,
                             bf16* __restrict__ out, int n) {
    int i = blockIdx.x * blockDim.x + threadIdx.x;
    if (i < n) out[i] = (bf16)in[i];
}

// fp32 [R][C] -> bf16 transposed [C][R] (weights; one-time, 1M elements each)
__global__ void cvt_f32_bf16_T(const float* __restrict__ in,
                               bf16* __restrict__ out, int R, int C) {
    int i = blockIdx.x * blockDim.x + threadIdx.x;
    if (i < R * C) {
        int r = i / C, c = i - r * C;
        out[(size_t)c * R + r] = (bf16)in[i];
    }
}

// ---------------------------------------------------------------------------
// Tiled bf16 GEMM: C[M,N] = scale * (A[M,K] @ B[K,N]) (+ bias),
// with B supplied TRANSPOSED: Bt[N][K] (so TDM dim-0 = K is contiguous and
// the LDS tile lands directly in B-fragment order [n][k]).
// Block: 256 threads = 8 waves; block tile 256(M) x 64(N); K step 32.
// Each wave owns a 32x64 strip: 2(M) x 4(N) accumulator tiles -> 8 WMMA/step.
// LDS B tile is double-buffered; wave 0 drives the Tensor Data Mover.
// mode 0: Cf[m*N+n] = acc*scale + bias[n]                (f32 out)
// mode 1: Cb[((b*H+h)*S+s)*DH + d] = bf16(acc*scale)     (Q/K layout)
// mode 2: Cb[((b*H+h)*DH+d)*S + s] = bf16(acc*scale)     (V transposed)
// ---------------------------------------------------------------------------
__global__ void __launch_bounds__(256)
gemm_bf16_wmma(const bf16* __restrict__ A, const bf16* __restrict__ Bt,
               int M, int N, int K, float scale,
               const float* __restrict__ bias,
               float* __restrict__ Cf, bf16* __restrict__ Cb, int mode) {
    __shared__ __align__(16) bf16 btile[2][64 * 32];   // [buf][n][k], 2 x 4 KB

    const int t    = threadIdx.x;
    const int wave = t >> 5;
    const int lane = t & 31;
    const int half = lane >> 4;
    const int l16  = lane & 15;

    const int mblk  = blockIdx.y * 256;
    const int nblk  = blockIdx.x * 64;
    const int steps = K >> 5;

    f32x8 acc[2][4] = {};

#if HAVE_TDM
    // Stage Bt[nblk:nblk+64, k0:k0+32] -> btile[buf] via Tensor Data Mover.
    // D# per CDNA5 ISA (2-D tensor, 2B elems).
    auto stage = [&](int buf, int k0) {
        if (wave == 0) {
            unsigned long long ga =
                (unsigned long long)(const void*)(Bt + (size_t)nblk * K + k0);
            unsigned int la = (unsigned int)(unsigned long long)(const void*)&btile[buf][0];
            u32x4 g0;
            g0[0] = 1u;                                   // count=1, user mode
            g0[1] = la;                                   // lds_addr
            g0[2] = (unsigned int)ga;                     // global_addr[31:0]
            g0[3] = (unsigned int)((ga >> 32) & 0x01FFFFFFu) | (2u << 30); // type=2
            i32x8 g1;
            g1[0] = (1 << 16);                            // mask=0, data_size=2B
            g1[1] = (int)((K & 0xFFFF) << 16);            // tensor_dim0 lo16 @ [63:48]
            g1[2] = (int)(((unsigned)K >> 16) & 0xFFFF) | ((N & 0xFFFF) << 16);
            g1[3] = (int)((((unsigned)N >> 16) & 0xFFFF) | (32u << 16)); // tile_dim0=32
            g1[4] = 64;                                   // tile_dim1=64, tile_dim2=0
            g1[5] = K;                                    // tensor_dim0_stride lo32
            g1[6] = 0;
            g1[7] = 0;
            i32x4 z4 = {0, 0, 0, 0};                      // groups 2/3: 2-D tensor
            i32x8 z8 = {0, 0, 0, 0, 0, 0, 0, 0};
            __builtin_amdgcn_tensor_load_to_lds(g0, g1, z4, z4, z8, 0);
        }
    };
#else
    // Fallback: cooperative load (coalesced along K since Bt is [N][K]).
    auto stage = [&](int buf, int k0) {
#pragma unroll
        for (int i = 0; i < 8; ++i) {
            int idx = i * 256 + t;
            int nn  = idx >> 5;       // 0..63
            int kk  = idx & 31;       // 0..31
            btile[buf][nn * 32 + kk] = Bt[(size_t)(nblk + nn) * K + k0 + kk];
        }
    };
#endif

    stage(0, 0);
    for (int i = 0; i < steps; ++i) {
        if (i + 1 < steps) stage((i + 1) & 1, (i + 1) * 32);   // overlap next DMA
#if HAVE_TDM
        if (wave == 0) {
            if (i + 1 < steps) WAIT_TENSORCNT(1);   // current tile complete
            else               WAIT_TENSORCNT(0);
        }
#endif
        __syncthreads();

        const int k0 = i * 32;
        // A fragments (16x32 bf16, documented lane layout), 2 M-subtiles
        Frag16 af[2];
#pragma unroll
        for (int mt = 0; mt < 2; ++mt) {
            const bf16* arow =
                A + (size_t)(mblk + wave * 32 + mt * 16 + l16) * K + k0;
            af[mt].h[0] = *(const bf16x8*)(arow + half * 8);      // K=half*8+0..7
            af[mt].h[1] = *(const bf16x8*)(arow + 16 + half * 8); // K=16+half*8+..
        }

        const bf16* bb = &btile[i & 1][0];
#pragma unroll
        for (int nt = 0; nt < 4; ++nt) {
            Frag16 bf_;
            const bf16* bcol = bb + (nt * 16 + l16) * 32 + half * 8;
            bf_.h[0] = *(const bf16x8*)(bcol);        // K = half*8 + 0..7
            bf_.h[1] = *(const bf16x8*)(bcol + 16);   // K = 16 + half*8 + 0..7
            acc[0][nt] = wmma_bf16(af[0].v, bf_.v, acc[0][nt]);
            acc[1][nt] = wmma_bf16(af[1].v, bf_.v, acc[1][nt]);
        }
        __syncthreads();
    }

    // ---- epilogue: C layout is VGPR r, lane L -> M=(L>>4)*8+r, N=L&15 ----
#pragma unroll
    for (int mt = 0; mt < 2; ++mt) {
#pragma unroll
        for (int nt = 0; nt < 4; ++nt) {
#pragma unroll
            for (int r = 0; r < 8; ++r) {
                int m = mblk + wave * 32 + mt * 16 + half * 8 + r;
                int n = nblk + nt * 16 + l16;
                float v = acc[mt][nt][r] * scale;
                if (mode == 0) {
                    Cf[(size_t)m * N + n] = v + (bias ? bias[n] : 0.0f);
                } else {
                    int b = m / SS, s = m - b * SS;
                    int h = n / DHD, d = n - h * DHD;
                    size_t off = (mode == 1)
                        ? (((size_t)(b * HH + h) * SS + s) * DHD + d)
                        : (((size_t)(b * HH + h) * DHD + d) * SS + s);
                    Cb[off] = (bf16)v;
                }
            }
        }
    }
}

// ---------------------------------------------------------------------------
// Flash-style attention: one wave per 16-query tile per (b,h).
// Q,K: [B][H][S][64] bf16 (Q pre-scaled by 1/8).  Vt: [B][H][64][S] bf16.
// Online softmax in f32; P converted C-layout -> A-layout via per-wave LDS.
// Output: Obf[b*S+s][h*64+d] bf16 (feeds the out-projection GEMM).
// ---------------------------------------------------------------------------
__global__ void __launch_bounds__(128)
attn_wmma(const bf16* __restrict__ Q, const bf16* __restrict__ Km,
          const bf16* __restrict__ Vt, bf16* __restrict__ Obf) {
    __shared__ __align__(16) bf16 pbuf[4][16 * 32];   // per-wave P tile, 4 KB

    const int t    = threadIdx.x;
    const int wave = t >> 5;
    const int lane = t & 31;
    const int half = lane >> 4;
    const int l16  = lane & 15;

    const int idx    = blockIdx.x * 4 + wave;    // global wave task
    const int qtiles = SS / 16;                  // 128
    const int bh     = idx / qtiles;
    const int qt     = idx - bh * qtiles;

    const bf16* q = Q  + (size_t)bh * SS * DHD;
    const bf16* k = Km + (size_t)bh * SS * DHD;
    const bf16* v = Vt + (size_t)bh * DHD * SS;
    bf16* pl = pbuf[wave];

    // Q fragments: A 16x64 -> two 16x32 fragments
    Frag16 qa[2];
    const bf16* qrow = q + (size_t)(qt * 16 + l16) * DHD;
#pragma unroll
    for (int kk = 0; kk < 2; ++kk) {
        qa[kk].h[0] = *(const bf16x8*)(qrow + kk * 32 + half * 8);
        qa[kk].h[1] = *(const bf16x8*)(qrow + kk * 32 + 16 + half * 8);
    }

    float run_max[8], run_sum[8];
#pragma unroll
    for (int r = 0; r < 8; ++r) { run_max[r] = -3.0e38f; run_sum[r] = 0.0f; }
    f32x8 oacc[4] = {};

    for (int kb = 0; kb < SS; kb += 32) {
        // prefetch next key/value tiles while this step computes
        if (kb + 32 < SS) {
            __builtin_prefetch(k + (size_t)(kb + 32 + l16) * DHD, 0, 3);
            __builtin_prefetch(v + (size_t)l16 * SS + kb + 32, 0, 3);
        }

        // ---- scores: S[16q x 32k] = Q(16x64) @ K^T(64x32) ----
        f32x8 s[2] = {};
#pragma unroll
        for (int st = 0; st < 2; ++st) {
            const bf16* krow = k + (size_t)(kb + st * 16 + l16) * DHD;
#pragma unroll
            for (int kk = 0; kk < 2; ++kk) {   // B elem e -> dh = kk*32+half*8+e (e<8), kk*32+16+half*8+(e-8)
                Frag16 bfr;
                bfr.h[0] = *(const bf16x8*)(krow + kk * 32 + half * 8);
                bfr.h[1] = *(const bf16x8*)(krow + kk * 32 + 16 + half * 8);
                s[st] = wmma_bf16(qa[kk].v, bfr.v, s[st]);
            }
        }

        // ---- online softmax (row = half*8+r spread across 16 lanes) ----
#pragma unroll
        for (int r = 0; r < 8; ++r) {
            float mx = fmaxf(s[0][r], s[1][r]);
#pragma unroll
            for (int off = 8; off; off >>= 1)
                mx = fmaxf(mx, __shfl_xor(mx, off, 16));
            float mnew = fmaxf(run_max[r], mx);
            float corr = __expf(run_max[r] - mnew);
            run_max[r] = mnew;
            float p0 = __expf(s[0][r] - mnew);
            float p1 = __expf(s[1][r] - mnew);
            float rs = p0 + p1;
#pragma unroll
            for (int off = 8; off; off >>= 1)
                rs += __shfl_xor(rs, off, 16);
            run_sum[r] = run_sum[r] * corr + rs;
#pragma unroll
            for (int nt = 0; nt < 4; ++nt) oacc[nt][r] *= corr;
            // C-layout -> LDS (row-major 16x32)
            int row = half * 8 + r;
            pl[row * 32 + l16]      = (bf16)p0;
            pl[row * 32 + 16 + l16] = (bf16)p1;
        }
        asm volatile("s_wait_dscnt 0" ::: "memory");   // DS RAW within wave

        // ---- P A-fragment from LDS, then O += P(16x32) @ V(32x64) ----
        Frag16 pa;
        {
            const bf16* prow = pl + l16 * 32;
            pa.h[0] = *(const bf16x8*)(prow + half * 8);
            pa.h[1] = *(const bf16x8*)(prow + 16 + half * 8);
        }
#pragma unroll
        for (int nt = 0; nt < 4; ++nt) {   // B elem e -> key = kb+half*8+e (e<8), kb+16+half*8+(e-8)
            Frag16 vb;
            const bf16* vrow = v + (size_t)(nt * 16 + l16) * SS + kb + half * 8;
            vb.h[0] = *(const bf16x8*)(vrow);
            vb.h[1] = *(const bf16x8*)(vrow + 16);
            oacc[nt] = wmma_bf16(pa.v, vb.v, oacc[nt]);
        }
    }

    // ---- normalize + store to [b*S+s][h*64+d] bf16 ----
    const int b = bh / HH, h = bh - b * HH;
#pragma unroll
    for (int r = 0; r < 8; ++r) {
        float inv = 1.0f / run_sum[r];
        int srow = qt * 16 + half * 8 + r;
        size_t base = ((size_t)b * SS + srow) * NPROJ + h * DHD;
#pragma unroll
        for (int nt = 0; nt < 4; ++nt)
            Obf[base + nt * 16 + l16] = (bf16)(oacc[nt][r] * inv);
    }
}

// ---------------------------------------------------------------------------
// Host-side orchestration
// ---------------------------------------------------------------------------
extern "C" void kernel_launch(void* const* d_in, const int* in_sizes, int n_in,
                              void* d_out, int out_size, void* d_ws, size_t ws_size,
                              hipStream_t stream) {
    const float* x  = (const float*)d_in[0];
    const float* Wq = (const float*)d_in[1];
    const float* Wk = (const float*)d_in[2];
    const float* Wv = (const float*)d_in[3];
    const float* Wo = (const float*)d_in[4];
    const float* bo = (const float*)d_in[5];

    const int nx = BB * SS * DD;      // 4,194,304
    const int nw = DD * NPROJ;        // 1,048,576

    char*  ws  = (char*)d_ws;
    size_t off = 0;
    auto carve = [&](size_t elems) {
        void* p = ws + off;
        off += (elems * sizeof(bf16) + 255) & ~(size_t)255;
        return (bf16*)p;
    };
    bf16* xbf   = carve(nx);
    bf16* wqt   = carve(nw);   // [NPROJ][DD]   (transposed)
    bf16* wkt   = carve(nw);
    bf16* wvt   = carve(nw);
    bf16* wot   = carve(nw);   // [DD][NPROJ]   (transposed)
    bf16* qbf   = carve(nx);   // [B][H][S][DH]
    bf16* kbf   = carve(nx);   // [B][H][S][DH]
    bf16* vtbf  = carve(nx);   // [B][H][DH][S]
    bf16* attbf = carve(nx);   // [B*S][H*DH]

    cvt_f32_bf16  <<<(nx + 255) / 256, 256, 0, stream>>>(x, xbf, nx);
    cvt_f32_bf16_T<<<(nw + 255) / 256, 256, 0, stream>>>(Wq, wqt, DD, NPROJ);
    cvt_f32_bf16_T<<<(nw + 255) / 256, 256, 0, stream>>>(Wk, wkt, DD, NPROJ);
    cvt_f32_bf16_T<<<(nw + 255) / 256, 256, 0, stream>>>(Wv, wvt, DD, NPROJ);
    cvt_f32_bf16_T<<<(nw + 255) / 256, 256, 0, stream>>>(Wo, wot, NPROJ, DD);

    dim3 gridG(NPROJ / 64, MTOT / 256);   // (16, 16)
    // Q/K/V projections (Q pre-scaled by 1/sqrt(DH))
    gemm_bf16_wmma<<<gridG, 256, 0, stream>>>(xbf, wqt, MTOT, NPROJ, DD,
                                              ATTN_SCALE, nullptr, nullptr, qbf, 1);
    gemm_bf16_wmma<<<gridG, 256, 0, stream>>>(xbf, wkt, MTOT, NPROJ, DD,
                                              1.0f, nullptr, nullptr, kbf, 1);
    gemm_bf16_wmma<<<gridG, 256, 0, stream>>>(xbf, wvt, MTOT, NPROJ, DD,
                                              1.0f, nullptr, nullptr, vtbf, 2);

    // attention: B*H*(S/16) = 4096 wave-tasks, 4 waves per block
    attn_wmma<<<(BB * HH * (SS / 16)) / 4, 128, 0, stream>>>(qbf, kbf, vtbf, attbf);

    // output projection + bias -> f32 d_out
    gemm_bf16_wmma<<<gridG, 256, 0, stream>>>(attbf, wot, MTOT, DD, NPROJ,
                                              1.0f, bo, (float*)d_out, nullptr, 0);
}
